// _GraphSAGETemporal_86053964742766
// MI455X (gfx1250) — matrix operand order repaired and checked by reading, not compiled, not run on hardware
//
#include <hip/hip_runtime.h>
#include <hip/hip_bf16.h>
#include <math.h>

#define TT 8
#define NN 50000
#define EE 800000
#define FF 128
#define HH 128
#define NHEADS_ 4
#define DH_ 32

typedef __attribute__((ext_vector_type(16))) _Float16 v16h;
typedef __attribute__((ext_vector_type(8)))  float    v8f;

// ---------------------------------------------------------------- gfx1250 async copy helpers
// GLOBAL_LOAD_ASYNC_TO_LDS_B128: per-lane 16B global->LDS DMA, tracked by ASYNCcnt.
__device__ __forceinline__ void async_g2l_b128(const void* gptr, void* lptr) {
  // Low 32 bits of a generic LDS address are the LDS byte offset (ISA 10.2 aperture rules).
  unsigned loff = (unsigned)(unsigned long long)(uintptr_t)lptr;
  asm volatile("global_load_async_to_lds_b128 %0, %1, off"
               :: "v"(loff), "v"(gptr) : "memory");
}
__device__ __forceinline__ void wait_async0() {
  asm volatile("s_wait_asynccnt 0" ::: "memory");
}

// ---------------------------------------------------------------- zero
__global__ __launch_bounds__(256) void k_zero(float* __restrict__ p, long n) {
  long i = (long)blockIdx.x * blockDim.x + threadIdx.x;
  if (i < n) p[i] = 0.0f;
}

// ---------------------------------------------------------------- weight pre-pack (f32 -> f16 WMMA B fragments)
// P[((ks*8 + ctile)*32 + lane)*16 + i] = W[k][n],
//   k = ks*32 + (i>>3)*16 + (lane>>4)*8 + (i&7),  n = ctile*16 + (lane&15)
__global__ __launch_bounds__(256) void k_pack_w(const float* __restrict__ W,
                                                _Float16* __restrict__ P) {
  int idx = blockIdx.x * 256 + threadIdx.x;       // 0..16383
  int i     = idx & 15;
  int lane  = (idx >> 4) & 31;
  int ctile = (idx >> 9) & 7;
  int ks    = idx >> 12;
  int k = ks * 32 + (i >> 3) * 16 + (lane >> 4) * 8 + (i & 7);
  int n = ctile * 16 + (lane & 15);
  P[idx] = (_Float16)W[k * HH + n];
}

// ---------------------------------------------------------------- scatter-add (segment sum + count)
__global__ __launch_bounds__(256) void k_scatter(
    const int* __restrict__ src, const int* __restrict__ dst,
    const float* __restrict__ x, float* __restrict__ agg, float* __restrict__ cnt)
{
  long gid = (long)blockIdx.x * blockDim.x + threadIdx.x;
  long e = gid >> 5;              // edge id
  int  g = (int)(gid & 31);       // channel group (4 floats each)
  if (e >= (long)EE) return;
  int s = src[e];
  int d = dst[e];
  float4 val = ((const float4*)(x + (size_t)s * HH))[g];
  float* ag = agg + (size_t)d * HH + g * 4;
  unsafeAtomicAdd(ag + 0, val.x);
  unsafeAtomicAdd(ag + 1, val.y);
  unsafeAtomicAdd(ag + 2, val.z);
  unsafeAtomicAdd(ag + 3, val.w);
  if (g == 0) unsafeAtomicAdd(cnt + d, 1.0f);
}

// ---------------------------------------------------------------- SAGE layer GEMM (WMMA, async-staged A)
// out[16 x 128] = relu( (agg/max(cnt,1)) @ Wl + x @ Wr + b )
// 256 threads = 8 waves; wave w owns output cols [16w, 16w+16).
// NOTE: relies on NN % 16 == 0 (50000 = 3125*16) so A tiles are full/contiguous.
__global__ __launch_bounds__(256) void k_sage_gemm(
    const float* __restrict__ agg, const float* __restrict__ cnt,
    const float* __restrict__ xin,
    const _Float16* __restrict__ WlP, const _Float16* __restrict__ WrP,
    const float* __restrict__ bias, float* __restrict__ out, int n_nodes)
{
  __shared__ float    sAf[16 * HH];   // raw aggregate tile (f32, async-staged)
  __shared__ float    sXf[16 * HH];   // raw self tile (f32, async-staged)
  __shared__ _Float16 sA16[16 * HH];  // scaled aggregate tile (f16)
  __shared__ _Float16 sX16[16 * HH];  // self tile (f16)
  __shared__ float    sInv[16];       // 1 / max(cnt,1) per row

  const int tid  = threadIdx.x;
  const int lane = tid & 31;
  const int wave = tid >> 5;
  const int row0 = blockIdx.x * 16;

  // --- async DMA the two contiguous 8KB A tiles into LDS (ASYNCcnt path) ---
  const float* aggT = agg + (size_t)row0 * HH;
  const float* xT   = xin + (size_t)row0 * HH;
  #pragma unroll
  for (int j = 0; j < 2; ++j) {
    int c16 = tid + j * 256;                 // 16-byte chunk index, 0..511
    async_g2l_b128(aggT + c16 * 4, &sAf[c16 * 4]);
    async_g2l_b128(xT   + c16 * 4, &sXf[c16 * 4]);
  }
  if (tid < 16) sInv[tid] = 1.0f / fmaxf(cnt[row0 + tid], 1.0f);
  wait_async0();
  __syncthreads();

  // --- convert to f16 with mean scaling ---
  for (int i = tid; i < 16 * HH; i += 256) {
    int r = i >> 7;
    sA16[i] = (_Float16)(sAf[i] * sInv[r]);
    sX16[i] = (_Float16)(sXf[i]);
  }
  __syncthreads();

  const int colBase = wave * 16;
  const int col     = colBase + (lane & 15);
  const int half    = lane >> 4;            // 0: lanes 0-15, 1: lanes 16-31
  const int r       = lane & 15;

  const v16h* WlF = (const v16h*)WlP;       // pre-packed fragments
  const v16h* WrF = (const v16h*)WrP;

  v8f acc = {};

  #pragma unroll
  for (int ks = 0; ks < 4; ++ks) {
    const int k0 = ks * 32;
    const int kb = k0 + half * 8;
    // A fragments (ISA 16-bit A 16x32 layout): two contiguous 16B chunks per lane.
    v16h aA, aX;
    #pragma unroll
    for (int i = 0; i < 8; ++i) {
      aA[i]     = sA16[r * HH + kb + i];
      aA[i + 8] = sA16[r * HH + kb + 16 + i];
      aX[i]     = sX16[r * HH + kb + i];
      aX[i + 8] = sX16[r * HH + kb + 16 + i];
    }
    // B fragments: one contiguous 32B load per lane from the packed weights.
    const int fidx = (ks * 8 + wave) * 32 + lane;
    v16h bL = WlF[fidx];
    v16h bR = WrF[fidx];

    acc = __builtin_amdgcn_wmma_f32_16x16x32_f16(false, aA, false, bL,
                                                 (short)0, acc, false, false);
    acc = __builtin_amdgcn_wmma_f32_16x16x32_f16(false, aX, false, bR,
                                                 (short)0, acc, false, false);
  }

  // D layout: VGPR j -> row j + 8*half, col = lane%16.
  const float bvv = bias[col];
  #pragma unroll
  for (int j = 0; j < 8; ++j) {
    int node = row0 + j + 8 * half;
    if (node < n_nodes) {
      float v = acc[j] + bvv;
      out[(size_t)node * HH + col] = fmaxf(v, 0.0f);
    }
  }
}

// ---------------------------------------------------------------- column mean over N nodes
__global__ __launch_bounds__(256) void k_colmean(
    const float* __restrict__ h, float* __restrict__ pooled, int n_nodes)
{
  __shared__ float red[256];
  const int c = blockIdx.x;
  float s = 0.0f;
  for (int n = threadIdx.x; n < n_nodes; n += 256)
    s += h[(size_t)n * HH + c];
  red[threadIdx.x] = s;
  __syncthreads();
  for (int off = 128; off > 0; off >>= 1) {
    if (threadIdx.x < off) red[threadIdx.x] += red[threadIdx.x + off];
    __syncthreads();
  }
  if (threadIdx.x == 0) pooled[c] = red[0] / (float)n_nodes;
}

// ---------------------------------------------------------------- attention + MLP head (tiny)
__global__ __launch_bounds__(128) void k_head(
    const float* __restrict__ pooled,
    const float* __restrict__ Wq, const float* __restrict__ bq,
    const float* __restrict__ Wk, const float* __restrict__ bk,
    const float* __restrict__ Wv, const float* __restrict__ bv,
    const float* __restrict__ Wo, const float* __restrict__ bo,
    const float* __restrict__ Wh1, const float* __restrict__ bh1,
    const float* __restrict__ Wh2, const float* __restrict__ bh2,
    float* __restrict__ out)
{
  __shared__ float sp[TT][HH];
  __shared__ float kk[TT][HH];
  __shared__ float vv[TT][HH];
  __shared__ float q7[HH];
  __shared__ float sraw[NHEADS_][TT];
  __shared__ float o7[HH];
  __shared__ float zz[HH];
  __shared__ float h1c[HH / 2];

  const int d = threadIdx.x;

  for (int t = 0; t < TT; ++t) sp[t][d] = pooled[t * HH + d];
  __syncthreads();

  for (int t = 0; t < TT; ++t) {
    float ksum = bk[d], vsum = bv[d];
    for (int j = 0; j < HH; ++j) {
      float pj = sp[t][j];
      ksum += pj * Wk[j * HH + d];
      vsum += pj * Wv[j * HH + d];
    }
    kk[t][d] = ksum;
    vv[t][d] = vsum;
  }
  {
    float qv = bq[d];
    for (int j = 0; j < HH; ++j) qv += sp[TT - 1][j] * Wq[j * HH + d];
    q7[d] = qv;
  }
  __syncthreads();

  if (d < NHEADS_ * TT) {
    int hh = d / TT, t = d % TT;
    float s = 0.0f;
    for (int i = 0; i < DH_; ++i) s += q7[hh * DH_ + i] * kk[t][hh * DH_ + i];
    sraw[hh][t] = s * 0.17677669529663687f;   // 1/sqrt(32)
  }
  __syncthreads();

  if (d < NHEADS_) {
    float m = -1e30f;
    for (int t = 0; t < TT; ++t) m = fmaxf(m, sraw[d][t]);
    float se = 0.0f;
    for (int t = 0; t < TT; ++t) { float e = expf(sraw[d][t] - m); sraw[d][t] = e; se += e; }
    float inv = 1.0f / se;
    for (int t = 0; t < TT; ++t) sraw[d][t] *= inv;
  }
  __syncthreads();

  {
    float o = 0.0f;
    int hh = d / DH_;
    for (int t = 0; t < TT; ++t) o += sraw[hh][t] * vv[t][d];
    o7[d] = o;
  }
  __syncthreads();

  {
    float z = bo[d];
    for (int j = 0; j < HH; ++j) z += o7[j] * Wo[j * HH + d];
    zz[d] = z;
  }
  __syncthreads();

  if (d < HH / 2) {
    float h = bh1[d];
    for (int j = 0; j < HH; ++j) h += zz[j] * Wh1[j * (HH / 2) + d];
    h1c[d] = fmaxf(h, 0.0f);
  }
  __syncthreads();

  if (d == 0) {
    float l = bh2[0];
    for (int m = 0; m < HH / 2; ++m) l += h1c[m] * Wh2[m];
    out[0] = 1.0f / (1.0f + expf(-l));
  }
}

// ---------------------------------------------------------------- host orchestration
extern "C" void kernel_launch(void* const* d_in, const int* in_sizes, int n_in,
                              void* d_out, int out_size, void* d_ws, size_t ws_size,
                              hipStream_t stream)
{
  (void)in_sizes; (void)n_in; (void)out_size; (void)ws_size;

  const float* xs  = (const float*)d_in[0];
  const int*   ei  = (const int*)  d_in[1];
  const float* Wl1 = (const float*)d_in[2];
  const float* Wr1 = (const float*)d_in[3];
  const float* b1  = (const float*)d_in[4];
  const float* Wl2 = (const float*)d_in[5];
  const float* Wr2 = (const float*)d_in[6];
  const float* b2  = (const float*)d_in[7];
  const float* Wl3 = (const float*)d_in[8];
  const float* Wr3 = (const float*)d_in[9];
  const float* b3  = (const float*)d_in[10];
  const float* Wq  = (const float*)d_in[11];
  const float* bq  = (const float*)d_in[12];
  const float* Wk  = (const float*)d_in[13];
  const float* bk  = (const float*)d_in[14];
  const float* Wv  = (const float*)d_in[15];
  const float* bv  = (const float*)d_in[16];
  const float* Wo  = (const float*)d_in[17];
  const float* bo  = (const float*)d_in[18];
  const float* Wh1 = (const float*)d_in[19];
  const float* bh1 = (const float*)d_in[20];
  const float* Wh2 = (const float*)d_in[21];
  const float* bh2 = (const float*)d_in[22];

  // Workspace (floats): agg[N*H] | cnt[N] | bufA[N*H] | bufB[N*H] | pooled[T*H] | wpack (6 x 16384 f16)
  float* agg    = (float*)d_ws;
  float* cnt    = agg  + (size_t)NN * HH;
  float* bufA   = cnt  + NN;
  float* bufB   = bufA + (size_t)NN * HH;
  float* pooled = bufB + (size_t)NN * HH;
  _Float16* wpack = (_Float16*)(pooled + TT * HH);   // 32B-aligned (offset % 32 == 0)

  const float* Wls[3] = {Wl1, Wl2, Wl3};
  const float* Wrs[3] = {Wr1, Wr2, Wr3};
  const float* bss[3] = {b1, b2, b3};

  _Float16* WlPk[3], * WrPk[3];
  for (int l = 0; l < 3; ++l) {
    WlPk[l] = wpack + (size_t)(2 * l + 0) * 16384;
    WrPk[l] = wpack + (size_t)(2 * l + 1) * 16384;
  }
  // One-time weight pre-pack into WMMA B-fragment order.
  for (int l = 0; l < 3; ++l) {
    k_pack_w<<<64, 256, 0, stream>>>(Wls[l], WlPk[l]);
    k_pack_w<<<64, 256, 0, stream>>>(Wrs[l], WrPk[l]);
  }

  const long zn = (long)NN * HH + NN;                 // agg + cnt contiguous
  const unsigned zgrid = (unsigned)((zn + 255) / 256);
  const unsigned sgrid = (unsigned)(((long)EE * 32 + 255) / 256);
  const unsigned ggrid = (unsigned)((NN + 15) / 16);

  for (int t = 0; t < TT; ++t) {
    const int*   src = ei + (size_t)t * 2 * EE;
    const int*   dst = src + EE;
    const float* in0 = xs + (size_t)t * NN * FF;

    const float* ins[3]  = {in0,  bufA, bufB};
    float*       outs[3] = {bufA, bufB, bufA};

    for (int l = 0; l < 3; ++l) {
      k_zero<<<zgrid, 256, 0, stream>>>(agg, zn);
      k_scatter<<<sgrid, 256, 0, stream>>>(src, dst, ins[l], agg, cnt);
      k_sage_gemm<<<ggrid, 256, 0, stream>>>(agg, cnt, ins[l], WlPk[l], WrPk[l],
                                             bss[l], outs[l], NN);
    }
    k_colmean<<<HH, 256, 0, stream>>>(bufA, pooled + t * HH, NN);
  }

  k_head<<<1, 128, 0, stream>>>(pooled, Wq, bq, Wk, bk, Wv, bv, Wo, bo,
                                Wh1, bh1, Wh2, bh2, (float*)d_out);
}
